// AttentionLayer_84851373899993
// MI455X (gfx1250) — matrix operand order repaired and checked by reading, not compiled, not run on hardware
//
#include <hip/hip_runtime.h>

typedef __attribute__((ext_vector_type(16))) __bf16 v16bf;
typedef __attribute__((ext_vector_type(8)))  __bf16 v8bf;
typedef __attribute__((ext_vector_type(4)))  __bf16 v4bf;
typedef __attribute__((ext_vector_type(8)))  float  v8f;

#define N_INST   256
#define BATCH    128
#define IN_DIM   1024
#define L_DIM    512
#define M_TOTAL  (N_INST * BATCH)      // 32768 rows
#define M_TILE   32                    // rows per workgroup
#define CHUNK_K  512                   // K staged in LDS per pass
#define KT_PER_CHUNK (CHUNK_K / 32)    // 16
#define KT_TOTAL (IN_DIM / 32)         // 32
#define NT_TOTAL (L_DIM / 16)          // 32
#define XS_STRIDE (CHUNK_K + 8)        // +16B pad: rotate LDS banks per row

// ---------------------------------------------------------------------------
// Kernel 1: convert v,u (f32) -> bf16, pre-packed into WMMA B-fragment layout.
// B (32x16 bf16): lanes 0-15 hold K=0..15 of the tile (half e -> K=e),
// lanes 16-31 hold K=16..31.  Fragment block = 32 lanes * 32 B,
// frag id = nt*KT_TOTAL + kt so the GEMM inner-kt loop streams contiguously.
// ---------------------------------------------------------------------------
__global__ __launch_bounds__(256) void prep_pack(const float* __restrict__ v,
                                                 const float* __restrict__ u,
                                                 __bf16* __restrict__ vb,
                                                 __bf16* __restrict__ ub) {
    const unsigned ME  = IN_DIM * L_DIM;                 // 524288
    unsigned gid = blockIdx.x * blockDim.x + threadIdx.x;
    if (gid >= 2u * ME) return;
    const float* src = (gid < ME) ? v  : u;
    __bf16*      dst = (gid < ME) ? vb : ub;
    unsigned e  = (gid < ME) ? gid : gid - ME;
    unsigned k  = e >> 9;            // row in [0,1024)
    unsigned n  = e & 511;           // col in [0,512)
    unsigned kt = k >> 5, kr = k & 31;
    unsigned nt = n >> 4, ln = n & 15;
    unsigned lane = ln | ((kr >> 4) << 4);
    unsigned krr  = kr & 15;
    unsigned frag = nt * KT_TOTAL + kt;
    unsigned off  = frag * 512u + lane * 16u + krr;      // in halves
    dst[off] = (__bf16)src[e];
}

__device__ __forceinline__ float fast_sigmoid(float x) {
    return 1.0f / (1.0f + __expf(-x));
}
__device__ __forceinline__ float fast_tanh(float x) {
    return 1.0f - 2.0f / (__expf(2.0f * x) + 1.0f);
}

// ---------------------------------------------------------------------------
// Kernel 2: fused  score = (tanh(x@v) * sigmoid(x@u)) @ w  for 32 rows per WG.
// 8 waves: rowgroup = wave>>2 (16 rows each), widx = wave&3 -> 8 N-tiles each.
// ---------------------------------------------------------------------------
__global__ __launch_bounds__(256) void gemm_score(const float*  __restrict__ x,
                                                  const __bf16* __restrict__ vb,
                                                  const __bf16* __restrict__ ub,
                                                  const float*  __restrict__ w,
                                                  float*        __restrict__ scores) {
    __shared__ __bf16 xs[M_TILE * XS_STRIDE];  // ~33 KB bf16 x tile (bank-padded)
    __shared__ float  part[4][M_TILE];         // deterministic cross-wave reduce

    const int tid      = threadIdx.x;
    const int lane     = tid & 31;
    const int wave     = tid >> 5;
    const int ln       = lane & 15;
    const int hi       = lane >> 4;
    const int rowgroup = wave >> 2;           // 0/1 -> rows 0..15 / 16..31
    const int widx     = wave & 3;            // which 8 N-tiles
    const int ntbase   = widx * 8;
    const int rowbase  = blockIdx.x * M_TILE;

    // One per-lane 64-bit base per matrix; per-nt stride (32768 B) and the
    // +16 B half-fragment offset fold into the 24-bit instruction offset.
    const __bf16* __restrict__ vptr =
        vb + (size_t)ntbase * KT_TOTAL * 512 + (unsigned)lane * 16u;
    const __bf16* __restrict__ uptr =
        ub + (size_t)ntbase * KT_TOTAL * 512 + (unsigned)lane * 16u;
    // Per-lane A-fragment base in LDS: row = rowgroup*16 + ln, K = hi*16.
    const __bf16* __restrict__ aptr =
        xs + (unsigned)(rowgroup * 16 + ln) * XS_STRIDE + (unsigned)hi * 16u;

    v8f accv[8], accu[8];
#pragma unroll
    for (int j = 0; j < 8; ++j) { accv[j] = (v8f)0.0f; accu[j] = (v8f)0.0f; }

    for (int kc = 0; kc < IN_DIM / CHUNK_K; ++kc) {
        __syncthreads();
        // cooperative f32 -> bf16 stage of the x tile (float4 per step)
        for (int i = tid * 4; i < M_TILE * CHUNK_K; i += 256 * 4) {
            const int row = i / CHUNK_K;
            const int kk  = i % CHUNK_K;
            const float4 f = *(const float4*)(x + (size_t)(rowbase + row) * IN_DIM
                                                + (size_t)kc * CHUNK_K + kk);
            v4bf b;
            b[0] = (__bf16)f.x; b[1] = (__bf16)f.y;
            b[2] = (__bf16)f.z; b[3] = (__bf16)f.w;
            *(v4bf*)(xs + row * XS_STRIDE + kk) = b;
        }
        __syncthreads();

#pragma unroll 4
        for (int ktl = 0; ktl < KT_PER_CHUNK; ++ktl) {
            const int kt = kc * KT_PER_CHUNK + ktl;
            const unsigned koff = (unsigned)kt * 512u;   // halves; runtime part
            // A fragment (16-bit A 16x32): halves 0..15 at K = hi*16 + e,
            // matching the packed B layout (two ds_load_b128)
            v8bf lo = *(const v8bf*)(aptr + ktl * 32);
            v8bf hh = *(const v8bf*)(aptr + ktl * 32 + 8);
            v16bf a = __builtin_shufflevector(lo, hh, 0, 1, 2, 3, 4, 5, 6, 7,
                                              8, 9, 10, 11, 12, 13, 14, 15);
#pragma unroll
            for (int j = 0; j < 8; ++j) {
                v16bf bvf = *(const v16bf*)(vptr + koff + j * (KT_TOTAL * 512));
                v16bf buf = *(const v16bf*)(uptr + koff + j * (KT_TOTAL * 512));
                accv[j] = __builtin_amdgcn_wmma_f32_16x16x32_bf16(
                    false, a, false, bvf, (short)0, accv[j], false, false);
                accu[j] = __builtin_amdgcn_wmma_f32_16x16x32_bf16(
                    false, a, false, buf, (short)0, accu[j], false, false);
            }
        }
    }

    // Epilogue: s[m] = sum_n tanh(h)*sigmoid(g)*w[n]
    // C/D layout: VGPR r holds (M = r + 8*hi, N = nt*16 + ln)
    float s[8];
#pragma unroll
    for (int r = 0; r < 8; ++r) s[r] = 0.0f;
#pragma unroll
    for (int j = 0; j < 8; ++j) {
        const float wn = w[(ntbase + j) * 16 + ln];
#pragma unroll
        for (int r = 0; r < 8; ++r) {
            const float t = fast_tanh(accv[j][r]);
            const float g = fast_sigmoid(accu[j][r]);
            s[r] += t * g * wn;
        }
    }
    // reduce over the 16 N-lanes within each half-wave
#pragma unroll
    for (int off = 1; off < 16; off <<= 1) {
#pragma unroll
        for (int r = 0; r < 8; ++r) s[r] += __shfl_xor(s[r], off, 32);
    }
    if (ln == 0) {
#pragma unroll
        for (int r = 0; r < 8; ++r)
            part[widx][rowgroup * 16 + hi * 8 + r] = s[r];
    }
    __syncthreads();
    if (tid < M_TILE) {
        scores[rowbase + tid] =
            part[0][tid] + part[1][tid] + part[2][tid] + part[3][tid];
    }
}

// ---------------------------------------------------------------------------
// Kernel 3: softmax over the instance axis (256 values) per batch column.
// ---------------------------------------------------------------------------
__global__ __launch_bounds__(256) void softmax_inst(const float* __restrict__ scores,
                                                    float* __restrict__ out) {
    __shared__ float red[256];
    const int b = blockIdx.x;       // batch column
    const int n = threadIdx.x;      // instance
    const float sc = scores[n * BATCH + b];

    red[n] = sc; __syncthreads();
    for (int off = 128; off > 0; off >>= 1) {
        if (n < off) red[n] = fmaxf(red[n], red[n + off]);
        __syncthreads();
    }
    const float mx = red[0];
    __syncthreads();

    const float e = __expf(sc - mx);
    red[n] = e; __syncthreads();
    for (int off = 128; off > 0; off >>= 1) {
        if (n < off) red[n] += red[n + off];
        __syncthreads();
    }
    out[n * BATCH + b] = e / red[0];
}

extern "C" void kernel_launch(void* const* d_in, const int* in_sizes, int n_in,
                              void* d_out, int out_size, void* d_ws, size_t ws_size,
                              hipStream_t stream) {
    const float* x = (const float*)d_in[0];   // [256,128,1024]
    const float* v = (const float*)d_in[1];   // [1024,512]
    const float* u = (const float*)d_in[2];   // [1024,512]
    const float* w = (const float*)d_in[3];   // [512,1]
    float* out = (float*)d_out;               // [256,128,1]

    char* ws = (char*)d_ws;
    __bf16* vb   = (__bf16*)ws;                                   // 1 MiB
    __bf16* ub   = vb + (size_t)IN_DIM * L_DIM;                   // 1 MiB
    float* scores = (float*)(ws + 2 * sizeof(__bf16) * (size_t)IN_DIM * L_DIM); // 128 KiB

    const unsigned packN = 2u * IN_DIM * L_DIM;
    prep_pack<<<(packN + 255) / 256, 256, 0, stream>>>(v, u, vb, ub);
    gemm_score<<<M_TOTAL / M_TILE, 256, 0, stream>>>(x, vb, ub, w, scores);
    softmax_inst<<<BATCH, 256, 0, stream>>>(scores, out);
}
